// conv2d_23545010716726
// MI455X (gfx1250) — matrix operand order repaired and checked by reading, not compiled, not run on hardware
//
#include <hip/hip_runtime.h>

// Hexagdly conv2d (size=1, stride=1, even width) as implicit GEMM on CDNA5 WMMA,
// with Tensor Data Mover (TDM) staging of the input tile into LDS.
// M = Cout(128), N = pixels, K = Cin(64) * 7 taps = 448 -> 14 x wmma_f32_16x16x32_f16.

typedef __attribute__((ext_vector_type(16))) _Float16     v16h;
typedef __attribute__((ext_vector_type(8)))  float        v8f;
typedef __attribute__((ext_vector_type(4)))  unsigned int uv4;
typedef __attribute__((ext_vector_type(8)))  int          iv8;
typedef __attribute__((ext_vector_type(4)))  int          iv4;

#define HH   256
#define WW   256
#define CIN  64
#define COUT 128
#define HP   258          // padded height (h' = h+1, zero border)
#define WP   258          // padded width  (w' = w+1, zero border)

// LDS input tile: 6 rows x 66 halo-cols x (32 ci-pairs + 4 pad) dwords
#define TROWS 6
#define TCOLS 66
#define CPAD  36
#define ROWDW (TCOLS * CPAD)   // 2376 dwords per tile row

#if defined(__has_builtin)
#if __has_builtin(__builtin_amdgcn_tensor_load_to_lds) && __has_builtin(__builtin_amdgcn_s_wait_tensorcnt)
#define HAVE_TDM 1
#endif
#endif
#ifndef HAVE_TDM
#define HAVE_TDM 0
#endif

// ---------------------------------------------------------------------------
// Pre-pack: x (f32 NCHW) -> padded f16-pair dwords xp[b][h'][w'][cp]
// (cp = ci/2; dword = {f16(x[2cp]), f16(x[2cp+1])}; borders h'/w' in {0,257} are zero)
// ---------------------------------------------------------------------------
__global__ __launch_bounds__(256)
void hex_prepack_kernel(const float* __restrict__ x, uint32_t* __restrict__ xp)
{
    __shared__ _Float16 lh[CIN * 65];            // 64 ci x 64 w tile, pad stride 65
    const int tid = threadIdx.x;
    const int w0  = blockIdx.x * 64;
    const int h   = blockIdx.y;
    const int b   = blockIdx.z;

    // Phase A: coalesced f32 reads along w, f16 convert into LDS
    for (int i = tid; i < CIN * 64; i += 256) {
        const int ci = i >> 6, wl = i & 63;
        lh[ci * 65 + wl] = (_Float16)x[((size_t)(b * CIN + ci) * HH + h) * WW + w0 + wl];
    }
    __syncthreads();

    // Phase B: transpose to [w][cp] pairs, coalesced 128B dword writes
    for (int i = tid; i < 64 * 32; i += 256) {
        const int cp = i & 31, wl = i >> 5;
        union { uint32_t u; _Float16 f[2]; } pk;
        pk.f[0] = lh[(2 * cp) * 65 + wl];
        pk.f[1] = lh[(2 * cp + 1) * 65 + wl];
        xp[(((size_t)b * HP + (h + 1)) * WP + (w0 + wl + 1)) * 32 + cp] = pk.u;
    }
}

// Zero the padded borders (h' in {0,257}, w' in {0,257}) so TDM tiles never need OOB.
__global__ __launch_bounds__(256)
void hex_border_zero_kernel(uint32_t* __restrict__ xp)
{
    const int perb = (2 * WP + 2 * (HP - 2)) * 32;   // 1028 positions * 32 cp
    int idx = blockIdx.x * 256 + threadIdx.x;
    if (idx >= 8 * perb) return;
    const int b   = idx / perb;
    const int rem = idx % perb;
    const int pos = rem >> 5, cp = rem & 31;
    int hq, wq;
    if (pos < 2 * WP) { hq = (pos < WP) ? 0 : (HP - 1); wq = pos % WP; }
    else { const int p2 = pos - 2 * WP;
           wq = (p2 < (HP - 2)) ? 0 : (WP - 1); hq = 1 + (p2 % (HP - 2)); }
    xp[(((size_t)b * HP + hq) * WP + wq) * 32 + cp] = 0u;
}

#if HAVE_TDM
// ---------------------------------------------------------------------------
// TDM: load a [nrows h'] x [66 w'] x [32 cp] dword tile into LDS, inserting
// 4 pad dwords after every 32 (pad_interval code 4 = 32 DW, pad_amount code 3 = 4 DW)
// -> LDS rows of 36 dwords, matching the conflict-free compute layout.
// ---------------------------------------------------------------------------
__device__ __forceinline__ void tdm_load_tile(const uint32_t* __restrict__ xp,
                                              uint32_t ldsoff,
                                              int b, int hp0, int wp0, int nrows)
{
    const uint64_t g = (uint64_t)(uintptr_t)xp +
        4ull * (((((uint64_t)b * HP) + (uint64_t)hp0) * WP + (uint64_t)wp0) * 32ull);
    uv4 g0;
    g0.x = 1u;                                   // count=1 (valid user descriptor)
    g0.y = ldsoff;                               // lds_addr (bytes)
    g0.z = (uint32_t)g;                          // global_addr[31:0]
    g0.w = (uint32_t)(g >> 32) | 0x80000000u;    // global_addr[56:32] | type=2
    iv8 g1;
    g1[0] = (int)((2u << 16)                     // data_size = 4B (dword units)
                | (1u << 20)                     // pad_enable
                | (4u << 22)                     // pad_interval: 32 dwords
                | (3u << 25));                   // pad_amount: 4 dwords
    g1[1] = (int)(32u << 16);                    // tensor_dim0 = 32 (cp)
    g1[2] = (int)(66u << 16);                    // tensor_dim1 = 66 (w')
    g1[3] = (int)(32u << 16);                    // tile_dim0 = 32
    g1[4] = (int)(66u | ((uint32_t)nrows << 16));// tile_dim1 = 66, tile_dim2 = nrows
    g1[5] = 32;                                  // tensor_dim0_stride (w' step, dwords)
    g1[6] = (int)((uint32_t)(WP * 32) << 16);    // tensor_dim1_stride[15:0] = 8256 (h' step)
    g1[7] = 0;                                   // tensor_dim1_stride[47:16] = 0
    iv4 g2; g2[0] = nrows; g2[1] = 0; g2[2] = 0; g2[3] = 0;  // tensor_dim2
    iv4 g3; g3[0] = 0; g3[1] = 0; g3[2] = 0; g3[3] = 0;
    iv8 g4; g4[0] = 0; g4[1] = 0; g4[2] = 0; g4[3] = 0;      // arg5 (uncharacterized) = 0
    g4[4] = 0; g4[5] = 0; g4[6] = 0; g4[7] = 0;
    __builtin_amdgcn_tensor_load_to_lds(g0, g1, g2, g3, g4, 0);
}

// ---------------------------------------------------------------------------
// Main kernel (TDM path): reads pre-packed f16 input.
// ---------------------------------------------------------------------------
__global__ __launch_bounds__(256)
void hexconv_wmma_tdm_kernel(const uint32_t* __restrict__ xp,
                             const float* __restrict__ k0,
                             const float* __restrict__ k1,
                             const float* __restrict__ bias,
                             float* __restrict__ out)
{
    __shared__ uint32_t xs[TROWS * ROWDW];   // 57,024 bytes

    const int tid   = threadIdx.x;
    const int lane  = tid & 31;
    const int wave  = tid >> 5;
    const int n     = lane & 15;
    const int khalf = lane >> 4;

    const int wbase = blockIdx.x * 64;
    const int h0    = blockIdx.y * 4;
    const int b     = blockIdx.z;

    // Kick off TDM first (wave 0 only; TDM ignores EXEC, wave-uniform branch).
    if (wave == 0) {
        const uint32_t xsoff =
            (uint32_t)(uintptr_t)(__attribute__((address_space(3))) uint32_t*)xs;
        // rows h' = h0..h0+3 (covers output rows r=0,1), then h0+4..h0+5
        tdm_load_tile(xp, xsoff,                   b, h0,     wbase, 4);
        tdm_load_tile(xp, xsoff + 4u * ROWDW * 4u, b, h0 + 4, wbase, 2);
    }

    // ---- A fragments: weights f32 -> f16, packed in contiguous k order (k = 16*khalf + s) ----
    const int m0    = wave * 16;
    const int corow = m0 + n;
    v16h afrag[14];
    #pragma unroll
    for (int t = 0; t < 7; ++t) {
      #pragma unroll
      for (int c = 0; c < 2; ++c) {
        union { uint32_t u[8]; v16h h; } au;
        #pragma unroll
        for (int v = 0; v < 8; ++v) {
          const int K  = (khalf << 4) + (v << 1);
          const int ci = c * 32 + K;
          float wa, wb;
          if (t < 3) {
            const int idx = (corow * CIN + ci) * 3 + t;
            wa = k0[idx]; wb = k0[idx + 3];
          } else {
            const int kh = (t - 3) >> 1, kw = (t - 3) & 1;
            const int idx = ((corow * CIN + ci) * 2 + kh) * 2 + kw;
            wa = k1[idx]; wb = k1[idx + 4];
          }
          union { uint32_t u; _Float16 f[2]; } pk;
          pk.f[0] = (_Float16)wa; pk.f[1] = (_Float16)wb;
          au.u[v] = pk.u;
        }
        afrag[t * 2 + c] = au.h;
      }
    }

    float bvals[8];
    #pragma unroll
    for (int e = 0; e < 8; ++e) bvals[e] = bias[m0 + khalf * 8 + e];

    const int parity = n & 1;

    auto do_row = [&](int r) {
      for (int j = 0; j < 4; ++j) {
        v8f acc = {};
        #pragma unroll
        for (int t = 0; t < 7; ++t) {
          int dh, dw;
          if (t < 3) { dh = t - 1; dw = 0; }
          else { const int kh = (t - 3) >> 1, kw = (t - 3) & 1;
                 dh = kh - 1 + parity; dw = 2 * kw - 1; }
          const int hrow    = r + 1 + dh;                    // 0..5
          const int wi      = j * 16 + n + 1 + dw;           // 0..65
          const int rowbase = (hrow * TCOLS + wi) * CPAD + khalf * 8;
          #pragma unroll
          for (int c = 0; c < 2; ++c) {
            union { uint4 q[2]; v16h h; } bu;
            const uint32_t* p = &xs[rowbase + c * 16];
            bu.q[0] = *(const uint4*)(p);      // ds_load_b128
            bu.q[1] = *(const uint4*)(p + 4);  // ds_load_b128
            acc = __builtin_amdgcn_wmma_f32_16x16x32_f16(
                false, afrag[t * 2 + c], false, bu.h, (short)0, acc, false, false);
          }
        }
        const int h = h0 + r;
        const int w = wbase + j * 16 + n;
        #pragma unroll
        for (int e = 0; e < 8; ++e) {
          const int co = m0 + khalf * 8 + e;
          out[((size_t)(b * COUT + co) * HH + h) * WW + w] = acc[e] + bvals[e];
        }
      }
    };

    if (wave == 0) __builtin_amdgcn_s_wait_tensorcnt(1);  // first 4 rows landed
    __syncthreads();
    do_row(0); do_row(1);
    if (wave == 0) __builtin_amdgcn_s_wait_tensorcnt(0);  // last 2 rows landed
    __syncthreads();
    do_row(2); do_row(3);
}
#endif  // HAVE_TDM

// ---------------------------------------------------------------------------
// Fallback (round-1 kernel): self-staging from f32, no workspace needed.
// ---------------------------------------------------------------------------
__global__ __launch_bounds__(256)
void hexconv_wmma_fallback_kernel(const float* __restrict__ x,
                                  const float* __restrict__ k0,
                                  const float* __restrict__ k1,
                                  const float* __restrict__ bias,
                                  float* __restrict__ out)
{
    __shared__ uint32_t xs[TROWS * ROWDW];

    const int tid   = threadIdx.x;
    const int lane  = tid & 31;
    const int wave  = tid >> 5;
    const int n     = lane & 15;
    const int khalf = lane >> 4;

    const int wbase = blockIdx.x * 64;
    const int h0    = blockIdx.y * 4;
    const int b     = blockIdx.z;

    const int m0    = wave * 16;
    const int corow = m0 + n;
    v16h afrag[14];
    #pragma unroll
    for (int t = 0; t < 7; ++t) {
      #pragma unroll
      for (int c = 0; c < 2; ++c) {
        union { uint32_t u[8]; v16h h; } au;
        #pragma unroll
        for (int v = 0; v < 8; ++v) {
          const int K  = (khalf << 4) + (v << 1);
          const int ci = c * 32 + K;
          float wa, wb;
          if (t < 3) {
            const int idx = (corow * CIN + ci) * 3 + t;
            wa = k0[idx]; wb = k0[idx + 3];
          } else {
            const int kh = (t - 3) >> 1, kw = (t - 3) & 1;
            const int idx = ((corow * CIN + ci) * 2 + kh) * 2 + kw;
            wa = k1[idx]; wb = k1[idx + 4];
          }
          union { uint32_t u; _Float16 f[2]; } pk;
          pk.f[0] = (_Float16)wa; pk.f[1] = (_Float16)wb;
          au.u[v] = pk.u;
        }
        afrag[t * 2 + c] = au.h;
      }
    }

    float bvals[8];
    #pragma unroll
    for (int e = 0; e < 8; ++e) bvals[e] = bias[m0 + khalf * 8 + e];

    for (int i = tid; i < TROWS * 32 * TCOLS; i += 256) {
      const int wi = i % TCOLS;
      const int cp = (i / TCOLS) & 31;
      const int hh = i / (TCOLS * 32);
      const int h  = h0 - 1 + hh;
      const int w  = wbase - 1 + wi;
      union { uint32_t u; _Float16 f[2]; } pk;
      pk.u = 0u;
      if ((unsigned)h < HH && (unsigned)w < WW) {
        const size_t base = ((size_t)(b * CIN + 2 * cp) * HH + h) * WW + w;
        pk.f[0] = (_Float16)x[base];
        pk.f[1] = (_Float16)x[base + (size_t)HH * WW];
      }
      xs[(hh * TCOLS + wi) * CPAD + cp] = pk.u;
    }
    __syncthreads();

    const int parity = n & 1;
    for (int r = 0; r < 4; ++r) {
      for (int j = 0; j < 4; ++j) {
        v8f acc = {};
        #pragma unroll
        for (int t = 0; t < 7; ++t) {
          int dh, dw;
          if (t < 3) { dh = t - 1; dw = 0; }
          else { const int kh = (t - 3) >> 1, kw = (t - 3) & 1;
                 dh = kh - 1 + parity; dw = 2 * kw - 1; }
          const int hrow    = r + 1 + dh;
          const int wi      = j * 16 + n + 1 + dw;
          const int rowbase = (hrow * TCOLS + wi) * CPAD + khalf * 8;
          #pragma unroll
          for (int c = 0; c < 2; ++c) {
            union { uint4 q[2]; v16h h; } bu;
            const uint32_t* p = &xs[rowbase + c * 16];
            bu.q[0] = *(const uint4*)(p);
            bu.q[1] = *(const uint4*)(p + 4);
            acc = __builtin_amdgcn_wmma_f32_16x16x32_f16(
                false, afrag[t * 2 + c], false, bu.h, (short)0, acc, false, false);
          }
        }
        const int h = h0 + r;
        const int w = wbase + j * 16 + n;
        #pragma unroll
        for (int e = 0; e < 8; ++e) {
          const int co = m0 + khalf * 8 + e;
          out[((size_t)(b * COUT + co) * HH + h) * WW + w] = acc[e] + bvals[e];
        }
      }
    }
}

extern "C" void kernel_launch(void* const* d_in, const int* in_sizes, int n_in,
                              void* d_out, int out_size, void* d_ws, size_t ws_size,
                              hipStream_t stream) {
    const float* x  = (const float*)d_in[0];   // (8,64,256,256) f32
    const float* k0 = (const float*)d_in[1];   // (128,64,3,1)   f32
    const float* k1 = (const float*)d_in[2];   // (128,64,2,2)   f32
    const float* bs = (const float*)d_in[3];   // (128,)         f32
    float* out = (float*)d_out;                // (8,128,256,256) f32
    (void)in_sizes; (void)n_in; (void)out_size;

    const size_t need = (size_t)8 * HP * WP * 32 * 4;   // 68,161,536 B

#if HAVE_TDM
    if (ws_size >= need) {
        uint32_t* xp = (uint32_t*)d_ws;
        const int border_elems = 8 * (2 * WP + 2 * (HP - 2)) * 32;
        hex_border_zero_kernel<<<(border_elems + 255) / 256, 256, 0, stream>>>(xp);
        hex_prepack_kernel<<<dim3(4, 256, 8), 256, 0, stream>>>(x, xp);
        hexconv_wmma_tdm_kernel<<<dim3(4, 64, 8), 256, 0, stream>>>(xp, k0, k1, bs, out);
        return;
    }
#endif
    (void)d_ws; (void)ws_size; (void)need;
    hexconv_wmma_fallback_kernel<<<dim3(4, 64, 8), 256, 0, stream>>>(x, k0, k1, bs, out);
}
